// MultiHeadAttention_38345468018779
// MI455X (gfx1250) — matrix operand order repaired and checked by reading, not compiled, not run on hardware
//
#include <hip/hip_runtime.h>

typedef __attribute__((ext_vector_type(16))) _Float16 v16h;
typedef __attribute__((ext_vector_type(8)))  _Float16 v8h;
typedef __attribute__((ext_vector_type(4)))  _Float16 v4h;
typedef __attribute__((ext_vector_type(8)))  float    v8f;
typedef __attribute__((ext_vector_type(4)))  float    v4f;

#define D_MODEL 1024
#define N_HEAD  16
#define DHEAD   64
#define BATCH   2
#define SEQ     2048
#define M_TOT   (BATCH * SEQ)

// ---------------------------------------------------------------------------
// WMMA fragment loaders (v_wmma_f32_16x16x32_f16; A 16x32, B 32x16, C/D 16x16)
//
// A layout (ISA 7.12.2, 16-bit A 16x32): lane L: row M=L&15, holds
// K={k0..k0+7} in v[0..7] and K={16+k0..23+k0} in v[8..15], k0=(L>=16)?8:0.
// B layout: lane L: col N=L&15, holds the same K order as A:
// K={k0..k0+7} in v[0..7] and K={16+k0..23+k0} in v[8..15].
// C/D layout: VGPR r: M = (lane>=16 ? 8 : 0) + r, N = lane & 15.
// ---------------------------------------------------------------------------

__device__ __forceinline__ v16h load_fragA(const _Float16* base, int ld) {
  const int lane = threadIdx.x & 31;
  const _Float16* p = base + (lane & 15) * ld + ((lane >> 4) << 3);
  v8h lo = *(const v8h*)(p);        // K k0..k0+7
  v8h hi = *(const v8h*)(p + 16);   // K k0+16..k0+23
  v16h r;
#pragma unroll
  for (int i = 0; i < 8; ++i) { r[i] = lo[i]; r[8 + i] = hi[i]; }
  return r;
}

__device__ __forceinline__ v16h load_fragA(const float* base, int ld) {
  const int lane = threadIdx.x & 31;
  const float* p = base + (lane & 15) * ld + ((lane >> 4) << 3);
  v4f a0 = *(const v4f*)(p);
  v4f a1 = *(const v4f*)(p + 4);
  v4f b0 = *(const v4f*)(p + 16);
  v4f b1 = *(const v4f*)(p + 20);
  v16h r;
#pragma unroll
  for (int i = 0; i < 4; ++i) {
    r[i]      = (_Float16)a0[i];
    r[4 + i]  = (_Float16)a1[i];
    r[8 + i]  = (_Float16)b0[i];
    r[12 + i] = (_Float16)b1[i];
  }
  return r;
}

// B operand: "base" points at row-major [N][K] storage of B^T (lane n reads
// row n = column n of B). ld = K-stride. Pure 16B loads, no conversion.
__device__ __forceinline__ v16h load_fragB(const _Float16* base, int ld) {
  const int lane = threadIdx.x & 31;
  const _Float16* p = base + (lane & 15) * ld + ((lane >> 4) << 3);
  v8h lo = *(const v8h*)(p);
  v8h hi = *(const v8h*)(p + 16);
  v16h r;
#pragma unroll
  for (int i = 0; i < 8; ++i) { r[i] = lo[i]; r[8 + i] = hi[i]; }
  return r;
}

__device__ __forceinline__ v8f wmma_f16(v16h a, v16h b, v8f c) {
  return __builtin_amdgcn_wmma_f32_16x16x32_f16(
      false, a, false, b, (short)0, c, false, false);
}

__device__ __forceinline__ v8f zero_v8f() {
  v8f z;
#pragma unroll
  for (int i = 0; i < 8; ++i) z[i] = 0.0f;
  return z;
}

// ---------------------------------------------------------------------------
// 16-lane butterfly reductions in pure VALU via DPP16 (no LDS, no waits).
// Steps: quad xor1 (0xB1), quad xor2 (0x4E), row_half_mirror (0x141),
// row_mirror (0x140). DPP rows are 16 lanes on wave32 == our C-frag group.
// ---------------------------------------------------------------------------
template <int CTRL>
__device__ __forceinline__ float dpp_xchg(float x) {
  int p = __builtin_amdgcn_mov_dpp(__float_as_int(x), CTRL, 0xF, 0xF, true);
  return __int_as_float(p);
}

__device__ __forceinline__ float red16_max(float x) {
  x = fmaxf(x, dpp_xchg<0xB1>(x));
  x = fmaxf(x, dpp_xchg<0x4E>(x));
  x = fmaxf(x, dpp_xchg<0x141>(x));
  x = fmaxf(x, dpp_xchg<0x140>(x));
  return x;
}

__device__ __forceinline__ float red16_sum(float x) {
  x += dpp_xchg<0xB1>(x);
  x += dpp_xchg<0x4E>(x);
  x += dpp_xchg<0x141>(x);
  x += dpp_xchg<0x140>(x);
  return x;
}

// ---------------------------------------------------------------------------
// One-shot f32 -> f16 convert (for weights): makes GEMM B-side pure b128 loads
// ---------------------------------------------------------------------------
__global__ __launch_bounds__(256) void cvt_f32_f16(
    const float* __restrict__ src, _Float16* __restrict__ dst, int n) {
  int i = (blockIdx.x * 256 + threadIdx.x) * 4;
  if (i + 3 < n) {
    v4f x = *(const v4f*)(src + i);
    v4h y;
#pragma unroll
    for (int t = 0; t < 4; ++t) y[t] = (_Float16)x[t];
    *(v4h*)(dst + i) = y;
  }
}

// ---------------------------------------------------------------------------
// Projection GEMM:  Y[m,n] = sum_k X[m,k] * W16[n,k] + bias[n]
// MODE 0: Q -> f16 [B,H,S,dh], scaled by 1/sqrt(dh)
// MODE 1: K -> f16 [B,H,S,dh]
// MODE 2: V -> f16 [B,H,dh,S]  (transposed; vectorized 16B epilogue stores)
// MODE 3: out -> f32 row-major [M, D]
// 128 threads (4 waves); block tile 64x64; wave w owns rows w*16..+15.
// ---------------------------------------------------------------------------
template <int MODE, typename XT>
__global__ __launch_bounds__(128) void proj_gemm(
    const XT* __restrict__ X, const _Float16* __restrict__ W,
    const float* __restrict__ bias, _Float16* __restrict__ outH,
    float* __restrict__ outF) {
  const int wave = threadIdx.x >> 5;
  const int lane = threadIdx.x & 31;
  const int m0 = blockIdx.x * 64 + wave * 16;
  const int n0 = blockIdx.y * 64;

  v8f acc[4];
#pragma unroll
  for (int j = 0; j < 4; ++j) acc[j] = zero_v8f();

  const XT* Xrow = X + m0 * D_MODEL;
  for (int kk = 0; kk < D_MODEL; kk += 32) {
    v16h a = load_fragA(Xrow + kk, D_MODEL);
#pragma unroll
    for (int j = 0; j < 4; ++j) {
      v16h b = load_fragB(W + (n0 + j * 16) * D_MODEL + kk, D_MODEL);
      acc[j] = wmma_f16(a, b, acc[j]);
    }
  }

  const int nl = lane & 15;
  const int mb = (lane >> 4) * 8;
#pragma unroll
  for (int j = 0; j < 4; ++j) {
    const int n = n0 + j * 16 + nl;
    const float bv = bias[n];
    const int h = n >> 6;   // n / 64
    const int dh = n & 63;  // n % 64
    if (MODE == 2) {
      // dh fixed per lane; rows m0+mb..+7 map to contiguous s -> one b128
      const int m = m0 + mb;
      const int bb = m >> 11, s = m & (SEQ - 1);
      v8h pack;
#pragma unroll
      for (int r = 0; r < 8; ++r) pack[r] = (_Float16)(acc[j][r] + bv);
      *(v8h*)(outH + ((bb * N_HEAD + h) * DHEAD + dh) * SEQ + s) = pack;
    } else {
#pragma unroll
      for (int r = 0; r < 8; ++r) {
        const int m = m0 + mb + r;
        const float val = acc[j][r] + bv;
        if (MODE == 0) {
          const int bb = m >> 11, s = m & (SEQ - 1);
          outH[((bb * N_HEAD + h) * SEQ + s) * DHEAD + dh] =
              (_Float16)(val * 0.125f);  // fold 1/sqrt(64)
        } else if (MODE == 1) {
          const int bb = m >> 11, s = m & (SEQ - 1);
          outH[((bb * N_HEAD + h) * SEQ + s) * DHEAD + dh] = (_Float16)val;
        } else {
          outF[m * D_MODEL + n] = val;
        }
      }
    }
  }
}

// ---------------------------------------------------------------------------
// Flash attention: grid (S/64 q-tiles, B*H). 4 waves, wave owns 16 q rows.
// Qh pre-scaled by 1/8. Kh: [B,H,S,dh]. Vt: [B,H,dh,S]. X2: f16 [B,S,H*dh].
// ---------------------------------------------------------------------------
__global__ __launch_bounds__(128) void attention_kernel(
    const _Float16* __restrict__ Qh, const _Float16* __restrict__ Kh,
    const _Float16* __restrict__ Vt, const int* __restrict__ mask,
    _Float16* __restrict__ X2) {
  __shared__ alignas(64) _Float16 lds_p[4][16 * 64];  // per-wave P staging

  const int wave = threadIdx.x >> 5;
  const int lane = threadIdx.x & 31;
  const int bh = blockIdx.y;
  const int bb = bh >> 4;
  const int h = bh & 15;
  const int q0 = blockIdx.x * 64 + wave * 16;

  const _Float16* Qbase = Qh + (bh * SEQ + q0) * DHEAD;
  const _Float16* Kbase = Kh + bh * SEQ * DHEAD;
  const _Float16* Vbase = Vt + bh * DHEAD * SEQ;
  const int* mbase = mask + (bb * SEQ + q0) * SEQ;

  const v16h qf0 = load_fragA(Qbase, DHEAD);
  const v16h qf1 = load_fragA(Qbase + 32, DHEAD);

  v8f o[4];
#pragma unroll
  for (int j = 0; j < 4; ++j) o[j] = zero_v8f();
  float row_max[8], row_sum[8];
#pragma unroll
  for (int r = 0; r < 8; ++r) { row_max[r] = -1e30f; row_sum[r] = 0.0f; }

  const int nl = lane & 15;
  const int mb = (lane >> 4) * 8;
  _Float16* pbuf = &lds_p[wave][0];

  for (int kt = 0; kt < SEQ; kt += 64) {
    if (kt + 64 < SEQ)
      __builtin_prefetch(Kbase + (kt + 64) * DHEAD, 0, 0);  // global_prefetch_b8

    // --- scores: S = (Q/8) @ K^T, 16 x 64 chunk -------------------------
    v8f s[4];
#pragma unroll
    for (int j = 0; j < 4; ++j) {
      v16h kf0 = load_fragB(Kbase + (kt + j * 16) * DHEAD, DHEAD);
      v16h kf1 = load_fragB(Kbase + (kt + j * 16) * DHEAD + 32, DHEAD);
      v8f z = wmma_f16(qf0, kf0, zero_v8f());
      s[j] = wmma_f16(qf1, kf1, z);
    }

    // --- mask + per-lane chunk row max ----------------------------------
    float cmax[8];
#pragma unroll
    for (int r = 0; r < 8; ++r) cmax[r] = -1e30f;
#pragma unroll
    for (int j = 0; j < 4; ++j) {
      const int n = kt + j * 16 + nl;
#pragma unroll
      for (int r = 0; r < 8; ++r) {
        float sv = s[j][r];
        sv = (mbase[(mb + r) * SEQ + n] == 0) ? -10000.0f : sv;
        s[j][r] = sv;
        cmax[r] = fmaxf(cmax[r], sv);
      }
    }
    // per-row max across the 16-lane group: pure-VALU DPP butterfly
#pragma unroll
    for (int r = 0; r < 8; ++r) cmax[r] = red16_max(cmax[r]);

    // --- online softmax rescale -----------------------------------------
#pragma unroll
    for (int r = 0; r < 8; ++r) {
      const float m_new = fmaxf(row_max[r], cmax[r]);
      const float alpha = __expf(row_max[r] - m_new);
      row_max[r] = m_new;
      row_sum[r] *= alpha;
#pragma unroll
      for (int j = 0; j < 4; ++j) o[j][r] *= alpha;
    }

    // --- P = exp(S - m); stage to LDS in row-major 16x64 ----------------
#pragma unroll
    for (int j = 0; j < 4; ++j) {
#pragma unroll
      for (int r = 0; r < 8; ++r) {
        const float p = __expf(s[j][r] - row_max[r]);
        row_sum[r] += p;  // per-lane partial, reduced at the end
        pbuf[(mb + r) * 64 + (j * 16 + nl)] = (_Float16)p;
      }
    }

    // --- O += P @ V ------------------------------------------------------
    v16h pa0 = load_fragA(pbuf, 64);       // seq kt..kt+31
    v16h pa1 = load_fragA(pbuf + 32, 64);  // seq kt+32..kt+63
#pragma unroll
    for (int jj = 0; jj < 4; ++jj) {
      v16h vb0 = load_fragB(Vbase + (jj * 16) * SEQ + kt, SEQ);
      v16h vb1 = load_fragB(Vbase + (jj * 16) * SEQ + kt + 32, SEQ);
      o[jj] = wmma_f16(pa0, vb0, o[jj]);
      o[jj] = wmma_f16(pa1, vb1, o[jj]);
    }
  }

  // final row-sum reduction across the 16-lane group (pure-VALU DPP)
#pragma unroll
  for (int r = 0; r < 8; ++r) row_sum[r] = red16_sum(row_sum[r]);

  // write normalized output: X2[b, s, h*64 + dh] (f16)
#pragma unroll
  for (int jj = 0; jj < 4; ++jj) {
    const int dh = jj * 16 + nl;
#pragma unroll
    for (int r = 0; r < 8; ++r) {
      const int srow = q0 + mb + r;
      const float val = o[jj][r] / row_sum[r];
      X2[(bb * SEQ + srow) * D_MODEL + h * DHEAD + dh] = (_Float16)val;
    }
  }
}

// ---------------------------------------------------------------------------
extern "C" void kernel_launch(void* const* d_in, const int* in_sizes, int n_in,
                              void* d_out, int out_size, void* d_ws,
                              size_t ws_size, hipStream_t stream) {
  (void)in_sizes; (void)n_in; (void)out_size; (void)ws_size;
  const float* q    = (const float*)d_in[0];
  const float* k    = (const float*)d_in[1];
  const float* v    = (const float*)d_in[2];
  const int*   mask = (const int*)d_in[3];
  const float* Wq   = (const float*)d_in[4];
  const float* bq   = (const float*)d_in[5];
  const float* Wk   = (const float*)d_in[6];
  const float* bk   = (const float*)d_in[7];
  const float* Wv   = (const float*)d_in[8];
  const float* bv   = (const float*)d_in[9];
  const float* Wo   = (const float*)d_in[10];
  const float* bo   = (const float*)d_in[11];
  float* out = (float*)d_out;

  char* ws = (char*)d_ws;
  const size_t SZ = (size_t)BATCH * N_HEAD * SEQ * DHEAD * sizeof(_Float16);  // 8 MB
  const size_t WSZ = (size_t)D_MODEL * D_MODEL * sizeof(_Float16);            // 2 MB
  _Float16* Qh   = (_Float16*)(ws);
  _Float16* Kh   = (_Float16*)(ws + SZ);
  _Float16* Vt   = (_Float16*)(ws + 2 * SZ);
  _Float16* X2   = (_Float16*)(ws + 3 * SZ);
  _Float16* Wq16 = (_Float16*)(ws + 4 * SZ);
  _Float16* Wk16 = (_Float16*)(ws + 4 * SZ + WSZ);
  _Float16* Wv16 = (_Float16*)(ws + 4 * SZ + 2 * WSZ);
  _Float16* Wo16 = (_Float16*)(ws + 4 * SZ + 3 * WSZ);

  const int wn = D_MODEL * D_MODEL;
  const int cvtb = wn / (256 * 4);
  cvt_f32_f16<<<cvtb, 256, 0, stream>>>(Wq, Wq16, wn);
  cvt_f32_f16<<<cvtb, 256, 0, stream>>>(Wk, Wk16, wn);
  cvt_f32_f16<<<cvtb, 256, 0, stream>>>(Wv, Wv16, wn);
  cvt_f32_f16<<<cvtb, 256, 0, stream>>>(Wo, Wo16, wn);

  dim3 gproj(M_TOT / 64, D_MODEL / 64);
  proj_gemm<0, float><<<gproj, 128, 0, stream>>>(q, Wq16, bq, Qh, nullptr);
  proj_gemm<1, float><<<gproj, 128, 0, stream>>>(k, Wk16, bk, Kh, nullptr);
  proj_gemm<2, float><<<gproj, 128, 0, stream>>>(v, Wv16, bv, Vt, nullptr);

  dim3 gattn(SEQ / 64, BATCH * N_HEAD);
  attention_kernel<<<gattn, 128, 0, stream>>>(Qh, Kh, Vt, mask, X2);

  proj_gemm<3, _Float16><<<gproj, 128, 0, stream>>>(X2, Wo16, bo, nullptr, out);
}
